// GroupedQueryAttention_86234353369554
// MI455X (gfx1250) — matrix operand order repaired and checked by reading, not compiled, not run on hardware
//
#include <hip/hip_runtime.h>
#include <hip/hip_bf16.h>

typedef __attribute__((ext_vector_type(16))) _Float16 v16h;
typedef __attribute__((ext_vector_type(8)))  _Float16 v8h;
typedef __attribute__((ext_vector_type(8)))  float    v8f;

#define S_LEN  2048
#define DMODEL 2048
#define DKV    512
#define DK     64
#define HQN    32

__device__ __forceinline__ v16h cat8(v8h lo, v8h hi) {
  return __builtin_shufflevector(lo, hi, 0,1,2,3,4,5,6,7,8,9,10,11,12,13,14,15);
}

// A-fragment (16x32 f16, ISA 7.12.2): lane l (r=l&15) holds row r,
// K = {ko..ko+7, ko+16..ko+23}, ko = (l>>4)*8. p -> row 0, k 0; row stride ldh halfs.
__device__ __forceinline__ v16h load_a_frag(const _Float16* p, int ldh) {
  int lane = threadIdx.x & 31;
  int r  = lane & 15;
  int ko = (lane >> 4) << 3;
  const _Float16* q = p + (size_t)r * ldh + ko;
  v8h lo = *(const v8h*)q;
  v8h hi = *(const v8h*)(q + 16);
  return cat8(lo, hi);
}

// B-fragment (32x16 f16): B stored as 16 rows (one per output column, K-contiguous),
// row stride ldh. Lane l (n=l&15) holds col n, K = {ko..ko+7, ko+16..ko+23}, ko = (l>>4)*8.
__device__ __forceinline__ v16h load_b_frag(const _Float16* p, int ldh) {
  int lane = threadIdx.x & 31;
  int r  = lane & 15;
  int ko = (lane >> 4) << 3;
  const _Float16* q = p + (size_t)r * ldh + ko;
  v8h lo = *(const v8h*)q;
  v8h hi = *(const v8h*)(q + 16);
  return cat8(lo, hi);
}

__device__ __forceinline__ v8f wmma_f16(v16h a, v16h b, v8f c) {
  return __builtin_amdgcn_wmma_f32_16x16x32_f16(false, a, false, b, (short)0, c, false, false);
}

__global__ __launch_bounds__(256) void cvt_f32_f16(const float* __restrict__ in,
                                                   _Float16* __restrict__ out, int n) {
  int i = blockIdx.x * 256 + threadIdx.x;
  if (i < n) out[i] = (_Float16)in[i];
}

// C = A[M,K] * W[N,K]^T + bias.
// 8 waves in a 2(M) x 4(N) grid; wave tile 64x32 (4 A-frags x 2 B-frags -> 8 WMMA/iter).
// Block tile 128x128. K loop unrolled 2x for load/WMMA overlap.
// mode 0: outF[m*N+n] (f32)                   (final O projection)
// mode 1: outH[((n>>6)*M + m)*64 + (n&63)]    (head-major Q / K)
// mode 2: outH[((n>>6)*64 + (n&63))*M + m]    (transposed V: [h][64][S])
__global__ __launch_bounds__(256) void gemm_qkvo(
    const _Float16* __restrict__ A, const _Float16* __restrict__ W,
    const float* __restrict__ bias,
    float* __restrict__ outF, _Float16* __restrict__ outH,
    int M, int N, int K, int mode)
{
  int wid  = threadIdx.x >> 5;
  int lane = threadIdx.x & 31;
  int m0 = blockIdx.x * 128 + (wid & 1) * 64;
  int n0 = blockIdx.y * 128 + (wid >> 1) * 32;

  v8f acc[4][2] = {};
  const _Float16* ap[4];
  const _Float16* bp[2];
  #pragma unroll
  for (int i = 0; i < 4; ++i) ap[i] = A + (size_t)(m0 + 16 * i) * K;
  #pragma unroll
  for (int j = 0; j < 2; ++j) bp[j] = W + (size_t)(n0 + 16 * j) * K;

  #pragma unroll 2
  for (int k = 0; k < K; k += 32) {
    v16h af[4], bf[2];
    #pragma unroll
    for (int i = 0; i < 4; ++i) af[i] = load_a_frag(ap[i] + k, K);
    #pragma unroll
    for (int j = 0; j < 2; ++j) bf[j] = load_b_frag(bp[j] + k, K);
    #pragma unroll
    for (int i = 0; i < 4; ++i)
      #pragma unroll
      for (int j = 0; j < 2; ++j)
        acc[i][j] = wmma_f16(af[i], bf[j], acc[i][j]);
  }

  int col  = lane & 15;
  int rowb = (lane >> 4) << 3;
  #pragma unroll
  for (int i = 0; i < 4; ++i) {
    #pragma unroll
    for (int j = 0; j < 2; ++j) {
      #pragma unroll
      for (int e = 0; e < 8; ++e) {
        int m = m0 + i * 16 + rowb + e;
        int n = n0 + j * 16 + col;
        float v = acc[i][j][e] + bias[n];
        if (mode == 0)      outF[(size_t)m * N + n] = v;
        else if (mode == 1) outH[(((size_t)(n >> 6) * M + m) << 6) + (n & 63)] = (_Float16)v;
        else                outH[((size_t)(n >> 6) * 64 + (n & 63)) * M + m]   = (_Float16)v;
      }
    }
  }
}

// Flash attention, causal. One wave = 16 query rows of one query head.
// Q: [32][S][64], Kc: [8][S][64], VT: [8][64][S], Oh: [S][2048] (f16).
__global__ __launch_bounds__(256) void attn_kernel(
    const _Float16* __restrict__ Q, const _Float16* __restrict__ Kc,
    const _Float16* __restrict__ VT, _Float16* __restrict__ Oh, int S)
{
  __shared__ _Float16 lds[8 * 16 * 32];
  int wid  = threadIdx.x >> 5;
  int lane = threadIdx.x & 31;
  int hq = blockIdx.y;
  int hk = hq >> 2;                        // GROUP = 4
  int qb = blockIdx.x * 128 + wid * 16;

  const _Float16* Qh = Q  + ((size_t)hq * S + qb) * DK;
  const _Float16* Kh = Kc + (size_t)hk * S * DK;
  const _Float16* Vh = VT + (size_t)hk * DK * S;
  _Float16* ldsw = lds + wid * (16 * 32);

  v16h qa0 = load_a_frag(Qh, DK);          // k 0..31
  v16h qa1 = load_a_frag(Qh + 32, DK);     // k 32..63

  v8f o0 = {}, o1 = {}, o2 = {}, o3 = {};
  float mrow[8], lrow[8];
  #pragma unroll
  for (int j = 0; j < 8; ++j) { mrow[j] = -1e30f; lrow[j] = 0.f; }

  int col  = lane & 15;
  int rowb = (lane >> 4) << 3;
  const float scale = 0.125f;              // 1/sqrt(64)

  for (int jb = 0; jb < qb + 16; jb += 32) {
    // S-tiles: Q(16x64) x K^T(64x32) -> two 16x16 f32 tiles
    v16h bk0  = load_b_frag(Kh + (size_t)jb        * DK,      DK);
    v16h bk0b = load_b_frag(Kh + (size_t)jb        * DK + 32, DK);
    v16h bk1  = load_b_frag(Kh + (size_t)(jb + 16) * DK,      DK);
    v16h bk1b = load_b_frag(Kh + (size_t)(jb + 16) * DK + 32, DK);
    v8f s0 = {}, s1 = {};
    s0 = wmma_f16(qa0, bk0,  s0);
    s0 = wmma_f16(qa1, bk0b, s0);
    s1 = wmma_f16(qa0, bk1,  s1);
    s1 = wmma_f16(qa1, bk1b, s1);

    int key0 = jb + col, key1 = jb + 16 + col;
    float rmax[8], psum[8];
    #pragma unroll
    for (int j = 0; j < 8; ++j) {
      int qrow = qb + rowb + j;
      float x0 = (key0 <= qrow) ? s0[j] * scale : -1e30f;
      float x1 = (key1 <= qrow) ? s1[j] * scale : -1e30f;
      s0[j] = x0; s1[j] = x1;
      rmax[j] = fmaxf(x0, x1);
    }
    #pragma unroll
    for (int d = 1; d < 16; d <<= 1)
      #pragma unroll
      for (int j = 0; j < 8; ++j)
        rmax[j] = fmaxf(rmax[j], __shfl_xor(rmax[j], d, 32));

    #pragma unroll
    for (int j = 0; j < 8; ++j) {
      float mnew = fmaxf(mrow[j], rmax[j]);
      float corr = __expf(mrow[j] - mnew);
      mrow[j] = mnew;
      float p0 = __expf(s0[j] - mnew);
      float p1 = __expf(s1[j] - mnew);
      s0[j] = p0; s1[j] = p1;
      psum[j] = p0 + p1;
      lrow[j] *= corr;
      o0[j] *= corr; o1[j] *= corr; o2[j] *= corr; o3[j] *= corr;
    }
    #pragma unroll
    for (int d = 1; d < 16; d <<= 1)
      #pragma unroll
      for (int j = 0; j < 8; ++j)
        psum[j] += __shfl_xor(psum[j], d, 32);
    #pragma unroll
    for (int j = 0; j < 8; ++j) lrow[j] += psum[j];

    // C-layout -> A-fragment layout via per-wave LDS tile (16x32 f16)
    #pragma unroll
    for (int j = 0; j < 8; ++j) {
      int m = rowb + j;
      ldsw[m * 32 + col]      = (_Float16)s0[j];
      ldsw[m * 32 + 16 + col] = (_Float16)s1[j];
    }
    __asm__ volatile("s_wait_dscnt 0" ::: "memory");
    v16h pa = load_a_frag(ldsw, 32);

    // O += P(16x32) x V(32x64); VT rows are K-contiguous B columns
    v16h bv0 = load_b_frag(Vh + (size_t)0  * S + jb, S);
    v16h bv1 = load_b_frag(Vh + (size_t)16 * S + jb, S);
    v16h bv2 = load_b_frag(Vh + (size_t)32 * S + jb, S);
    v16h bv3 = load_b_frag(Vh + (size_t)48 * S + jb, S);
    o0 = wmma_f16(pa, bv0, o0);
    o1 = wmma_f16(pa, bv1, o1);
    o2 = wmma_f16(pa, bv2, o2);
    o3 = wmma_f16(pa, bv3, o3);
  }

  #pragma unroll
  for (int j = 0; j < 8; ++j) {
    float inv = 1.0f / lrow[j];
    size_t rbase = (size_t)(qb + rowb + j) * DMODEL + (size_t)hq * DK;
    Oh[rbase + 0  + col] = (_Float16)(o0[j] * inv);
    Oh[rbase + 16 + col] = (_Float16)(o1[j] * inv);
    Oh[rbase + 32 + col] = (_Float16)(o2[j] * inv);
    Oh[rbase + 48 + col] = (_Float16)(o3[j] * inv);
  }
}

extern "C" void kernel_launch(void* const* d_in, const int* in_sizes, int n_in,
                              void* d_out, int out_size, void* d_ws, size_t ws_size,
                              hipStream_t stream) {
  const float* x  = (const float*)d_in[0];
  const float* wq = (const float*)d_in[1];
  const float* bq = (const float*)d_in[2];
  const float* wk = (const float*)d_in[3];
  const float* bk = (const float*)d_in[4];
  const float* wv = (const float*)d_in[5];
  const float* bv = (const float*)d_in[6];
  const float* wo = (const float*)d_in[7];
  const float* bo = (const float*)d_in[8];
  float* out = (float*)d_out;

  _Float16* ws = (_Float16*)d_ws;
  size_t p = 0;
  _Float16* xh  = ws + p; p += (size_t)S_LEN * DMODEL;   // 8 MB
  _Float16* wqh = ws + p; p += (size_t)DMODEL * DMODEL;  // 8 MB
  _Float16* wkh = ws + p; p += (size_t)DKV * DMODEL;     // 2 MB
  _Float16* wvh = ws + p; p += (size_t)DKV * DMODEL;     // 2 MB
  _Float16* woh = ws + p; p += (size_t)DMODEL * DMODEL;  // 8 MB
  _Float16* qh  = ws + p; p += (size_t)HQN * S_LEN * DK; // 8 MB
  _Float16* kh  = ws + p; p += (size_t)8 * S_LEN * DK;   // 2 MB
  _Float16* vth = ws + p; p += (size_t)8 * DK * S_LEN;   // 2 MB
  _Float16* ath = ws + p; p += (size_t)S_LEN * DMODEL;   // 8 MB

  int n1 = S_LEN * DMODEL;   // 4194304
  int n2 = DKV * DMODEL;     // 1048576
  cvt_f32_f16<<<(n1 + 255) / 256, 256, 0, stream>>>(x,  xh,  n1);
  cvt_f32_f16<<<(n1 + 255) / 256, 256, 0, stream>>>(wq, wqh, n1);
  cvt_f32_f16<<<(n2 + 255) / 256, 256, 0, stream>>>(wk, wkh, n2);
  cvt_f32_f16<<<(n2 + 255) / 256, 256, 0, stream>>>(wv, wvh, n2);
  cvt_f32_f16<<<(n1 + 255) / 256, 256, 0, stream>>>(wo, woh, n1);

  dim3 blk(256);
  gemm_qkvo<<<dim3(16, 16), blk, 0, stream>>>(xh, wqh, bq, nullptr, qh,  S_LEN, DMODEL, DMODEL, 1);
  gemm_qkvo<<<dim3(16, 4),  blk, 0, stream>>>(xh, wkh, bk, nullptr, kh,  S_LEN, DKV,    DMODEL, 1);
  gemm_qkvo<<<dim3(16, 4),  blk, 0, stream>>>(xh, wvh, bv, nullptr, vth, S_LEN, DKV,    DMODEL, 2);
  attn_kernel<<<dim3(16, 32), blk, 0, stream>>>(qh, kh, vth, ath, S_LEN);
  gemm_qkvo<<<dim3(16, 16), blk, 0, stream>>>(ath, woh, bo, out, nullptr, S_LEN, DMODEL, DMODEL, 0);
}
